// VectorQuantizer_3564822856192
// MI455X (gfx1250) — matrix operand order repaired and checked by reading, not compiled, not run on hardware
//
#include <hip/hip_runtime.h>
#include <hip/hip_bf16.h>
#include <math.h>

// ---------------------------------------------------------------------------
// VQ-VAE vector quantizer for MI455X (gfx1250, wave32, WMMA).
// Core: 16384 x 8192 x 512 f16 GEMM via v_wmma_f32_16x16x32_f16.
// emb tiles (64 codes x 512) staged global->LDS with ASYNC loads (ASYNCcnt),
// double-buffered and shared by all 8 waves; WMMA operands come from LDS via
// short-latency ds_load_b128, so no WMMA ever waits on L2 latency.
// argmax of score = x.(8192 e_k) - 4096||e_k||^2  ==  argmin distance.
// ---------------------------------------------------------------------------

typedef __attribute__((ext_vector_type(16))) _Float16 v16h;
typedef __attribute__((ext_vector_type(8)))  _Float16 v8h;
typedef __attribute__((ext_vector_type(8)))  float    v8f;
typedef int v4i_ __attribute__((vector_size(16)));   // matches builtin proto

#define NROWS   16384      // 32*512 flattened x rows
#define KCODES  8192
#define DDIM    512
#define QELEMS  8388608    // NROWS * DDIM
#define EMB_SCALE 8192.0f

#define XSTRIDE 520        // padded LDS row stride (halves): 4-bank lane skew
#define BSTRIDE 520
#define A_HALVES (8 * 16 * XSTRIDE)          // 66,560 halves (130 KB)
#define BBUF_HALVES (64 * BSTRIDE)           // 33,280 halves per buffer
#define LDS_BYTES ((A_HALVES + 2 * BBUF_HALVES) * 2)   // 266,240 B

#if __has_builtin(__builtin_amdgcn_global_load_async_to_lds_b128) && \
    __has_builtin(__builtin_amdgcn_s_wait_asynccnt)
#define HAVE_ASYNC 1
#else
#define HAVE_ASYNC 0
#endif

// ---------------- Kernel 1: convert x -> f16 -------------------------------
__global__ void vq_conv_x(const float* __restrict__ x, _Float16* __restrict__ xh) {
    int i = blockIdx.x * 256 + threadIdx.x;   // grid sized exactly
    xh[i] = (_Float16)x[i];
}

// ---------------- Kernel 2: convert emb -> f16*8192, bias = 4096*||e||^2 ---
__global__ void vq_conv_emb(const float* __restrict__ emb,
                            _Float16* __restrict__ eh,
                            float* __restrict__ cvec) {
    int k = blockIdx.x;            // one code per block
    int t = threadIdx.x;           // 256 threads, 2 elems each
    const float* row = emb + (size_t)k * DDIM;
    float v0 = row[t];
    float v1 = row[t + 256];
    eh[(size_t)k * DDIM + t]       = (_Float16)(v0 * EMB_SCALE);
    eh[(size_t)k * DDIM + t + 256] = (_Float16)(v1 * EMB_SCALE);
    __shared__ float red[256];
    red[t] = v0 * v0 + v1 * v1;
    __syncthreads();
    for (int o = 128; o > 0; o >>= 1) {
        if (t < o) red[t] += red[t + o];
        __syncthreads();
    }
    if (t == 0) cvec[k] = (EMB_SCALE * 0.5f) * red[0];   // 4096 * ||e_k||^2
}

// ---------------- Kernel 3: init scalars -----------------------------------
__global__ void vq_init(float* sumsq, int* maxidx) {
    if (threadIdx.x == 0) { *sumsq = 0.0f; *maxidx = 0; }
}

// ---------------- Kernel 4: WMMA distances + argmax ------------------------
__global__ void __launch_bounds__(256)
vq_argmin(const _Float16* __restrict__ xh,
          const _Float16* __restrict__ eh,
          const float* __restrict__ cvec,
          int* __restrict__ indices) {
    extern __shared__ _Float16 xs[];
    const int tid  = threadIdx.x;
    const int wave = tid >> 5;
    const int lane = tid & 31;
    const int half = lane >> 4;     // 0 or 1
    const int l16  = lane & 15;

    const int m0 = (blockIdx.x * 8 + wave) * 16;      // 16-row tile base
    _Float16* xw    = xs + (size_t)wave * 16 * XSTRIDE;  // wave's x-tile
    _Float16* bbase = xs + A_HALVES;                     // B double buffer

    // stage the wave's 16x512 x-tile into LDS (wave-private; same-wave DS
    // ordering guarantees RAW safety for our own later reads)
#pragma unroll
    for (int i = 0; i < 16; ++i) {
        int c   = i * 32 + lane;         // 32B chunk id; 32 chunks per row
        int r   = c >> 5;
        int col = (c & 31) * 16;         // halves
        v16h v = *(const v16h*)(xh + (size_t)(m0 + r) * DDIM + col);
        *(v16h*)(xw + (size_t)r * XSTRIDE + col) = v;
    }

    // stage a 64-code emb tile (64 KB) into LDS buffer dst_
#if HAVE_ASYNC
#define STAGE(dst_, c0_) do {                                                 \
        _Pragma("unroll")                                                     \
        for (int i_ = 0; i_ < 16; ++i_) {                                     \
            int ch_ = i_ * 256 + tid;        /* 16B chunk id (4096 total) */  \
            int r_  = ch_ >> 6;                                               \
            int c_  = (ch_ & 63) * 8;        /* halves */                     \
            __builtin_amdgcn_global_load_async_to_lds_b128(                   \
                (__attribute__((address_space(1))) v4i_*)                     \
                    (eh + (size_t)((c0_) + r_) * DDIM + c_),                  \
                (__attribute__((address_space(3))) v4i_*)                     \
                    ((dst_) + (size_t)r_ * BSTRIDE + c_),                     \
                0, 0);                                                        \
        }                                                                     \
    } while (0)
#define STAGE_WAIT() __builtin_amdgcn_s_wait_asynccnt(0)
#else
#define STAGE(dst_, c0_) do {                                                 \
        _Pragma("unroll")                                                     \
        for (int i_ = 0; i_ < 8; ++i_) {                                      \
            int ch_ = i_ * 256 + tid;        /* 32B chunk id (2048 total) */  \
            int r_  = ch_ >> 5;                                               \
            int c_  = (ch_ & 31) * 16;       /* halves */                     \
            *(v16h*)((dst_) + (size_t)r_ * BSTRIDE + c_) =                    \
                *(const v16h*)(eh + (size_t)((c0_) + r_) * DDIM + c_);        \
        }                                                                     \
    } while (0)
#define STAGE_WAIT() ((void)0)
#endif

    // A fragment: row l16, K chunks at half*8 and 16+half*8 (ISA 7.12.2)
    const _Float16* as = xw + (size_t)l16 * XSTRIDE + half * 8;
#define LDA(j_) __builtin_shufflevector(                                      \
        *(const v8h*)(as + (j_) * 32), *(const v8h*)(as + (j_) * 32 + 16),    \
        0,1,2,3,4,5,6,7,8,9,10,11,12,13,14,15)
    // B fragment: code row (t*16+l16), K chunks at half*8 and 16+half*8
#define LDB(cur_, t_, j_) __builtin_shufflevector(                            \
        *(const v8h*)((cur_) + ((t_) * 16 + l16) * BSTRIDE + (j_) * 32        \
                      + half * 8),                                            \
        *(const v8h*)((cur_) + ((t_) * 16 + l16) * BSTRIDE + (j_) * 32        \
                      + half * 8 + 16),                                       \
        0,1,2,3,4,5,6,7,8,9,10,11,12,13,14,15)

    float best[8];
    int   bidx[8];
#pragma unroll
    for (int i = 0; i < 8; ++i) { best[i] = -3.4e38f; bidx[i] = 0; }

    // prime: stage code tile 0 into buffer 0
    STAGE(bbase, 0);
    STAGE_WAIT();
    __syncthreads();

    for (int kt = 0; kt < KCODES / 64; ++kt) {
        const int c0 = kt * 64;
        _Float16* cur = bbase + (size_t)(kt & 1) * BBUF_HALVES;
        _Float16* nxt = bbase + (size_t)((kt + 1) & 1) * BBUF_HALVES;

        if (kt + 1 < KCODES / 64) STAGE(nxt, c0 + 64);   // overlaps compute

        // bias values: issued early, consumed only at the tile epilogue
        const float cv0 = cvec[c0 +  0 + l16];
        const float cv1 = cvec[c0 + 16 + l16];
        const float cv2 = cvec[c0 + 32 + l16];
        const float cv3 = cvec[c0 + 48 + l16];

        v8f acc0 = {}, acc1 = {}, acc2 = {}, acc3 = {};

#pragma unroll
        for (int j = 0; j < DDIM / 32; ++j) {
            v16h a  = LDA(j);             // LICM keeps these register-resident
            v16h b0 = LDB(cur, 0, j);
            v16h b1 = LDB(cur, 1, j);
            v16h b2 = LDB(cur, 2, j);
            v16h b3 = LDB(cur, 3, j);
            acc0 = __builtin_amdgcn_wmma_f32_16x16x32_f16(false, a, false, b0,
                        (short)0, acc0, false, false);
            acc1 = __builtin_amdgcn_wmma_f32_16x16x32_f16(false, a, false, b1,
                        (short)0, acc1, false, false);
            acc2 = __builtin_amdgcn_wmma_f32_16x16x32_f16(false, a, false, b2,
                        (short)0, acc2, false, false);
            acc3 = __builtin_amdgcn_wmma_f32_16x16x32_f16(false, a, false, b3,
                        (short)0, acc3, false, false);
        }

        // running per-lane argmax with bias applied here (codes ascend ->
        // strict '>' keeps the lowest index, matching argmin's first hit)
        const int cc0 = c0 +  0 + l16;
        const int cc1 = c0 + 16 + l16;
        const int cc2 = c0 + 32 + l16;
        const int cc3 = c0 + 48 + l16;
#pragma unroll
        for (int i = 0; i < 8; ++i) {
            float v;
            v = acc0[i] - cv0; if (v > best[i]) { best[i] = v; bidx[i] = cc0; }
            v = acc1[i] - cv1; if (v > best[i]) { best[i] = v; bidx[i] = cc1; }
            v = acc2[i] - cv2; if (v > best[i]) { best[i] = v; bidx[i] = cc2; }
            v = acc3[i] - cv3; if (v > best[i]) { best[i] = v; bidx[i] = cc3; }
        }

        // next tile fully staged + every wave done reading 'cur'
        STAGE_WAIT();
        __syncthreads();
    }
#undef STAGE
#undef STAGE_WAIT
#undef LDA
#undef LDB

    // reduce across the 16 lanes of each half (lane = code column)
#pragma unroll
    for (int off = 8; off >= 1; off >>= 1) {
#pragma unroll
        for (int i = 0; i < 8; ++i) {
            float ov = __shfl_xor(best[i], off, 32);
            int   oi = __shfl_xor(bidx[i], off, 32);
            if (ov > best[i] || (ov == best[i] && oi < bidx[i])) {
                best[i] = ov; bidx[i] = oi;
            }
        }
    }
    if (l16 == 0) {   // lane 0 -> rows m0..m0+7 ; lane 16 -> rows m0+8..m0+15
#pragma unroll
        for (int i = 0; i < 8; ++i)
            indices[m0 + half * 8 + i] = bidx[i];
    }
}

// ---------------- Kernel 5: gather + loss + max-index ----------------------
__global__ void vq_gather(const float* __restrict__ x,
                          const float* __restrict__ emb,
                          const int* __restrict__ indices,
                          float* __restrict__ qout,
                          float* __restrict__ idxout,
                          float* __restrict__ sumsq,
                          int* __restrict__ maxidx) {
    int n = blockIdx.x;       // one x-row per block
    int t = threadIdx.x;      // 256 threads, 2 elems each
    int idx = indices[n];
    const float* e  = emb + (size_t)idx * DDIM;
    const float* xr = x   + (size_t)n   * DDIM;
    float* o = qout + (size_t)n * DDIM;
    float v0 = e[t], v1 = e[t + 256];
    float d0 = v0 - xr[t], d1 = v1 - xr[t + 256];
    o[t] = v0;                // quantized_st == quantized numerically
    o[t + 256] = v1;
    __shared__ float red[256];
    red[t] = d0 * d0 + d1 * d1;
    __syncthreads();
    for (int o2 = 128; o2 > 0; o2 >>= 1) {
        if (t < o2) red[t] += red[t + o2];
        __syncthreads();
    }
    if (t == 0) {
        atomicAdd(sumsq, red[0]);
        atomicMax(maxidx, idx);
        idxout[n] = (float)idx;
    }
}

// ---------------- Kernel 6: finalize scalars -------------------------------
__global__ void vq_finalize(const float* __restrict__ sumsq,
                            const int* __restrict__ maxidx,
                            float* __restrict__ loss_out,
                            float* __restrict__ perp_out) {
    float mean = *sumsq / (float)QELEMS;
    loss_out[0] = 1.25f * mean;               // q_loss + 0.25*e_loss
    float L = (float)(*maxidx + 1);
    float p = 1.0f / L;
    perp_out[0] = expf(-p * logf(p + 1e-10f));
}

// ---------------------------------------------------------------------------
extern "C" void kernel_launch(void* const* d_in, const int* in_sizes, int n_in,
                              void* d_out, int out_size, void* d_ws, size_t ws_size,
                              hipStream_t stream) {
    const float* x   = (const float*)d_in[0];   // [16384, 512]
    const float* emb = (const float*)d_in[1];   // [8192, 512]
    float* out = (float*)d_out;

    // workspace layout
    char* ws = (char*)d_ws;
    const size_t MB = 1u << 20;
    _Float16* xh  = (_Float16*)(ws);                       // 16 MB
    _Float16* eh  = (_Float16*)(ws + 16 * MB);             //  8 MB
    float*    cvec = (float*)(ws + 24 * MB);               // 32 KB
    int*      indices = (int*)(ws + 24 * MB + (32u << 10));// 64 KB
    float*    sumsq   = (float*)(ws + 24 * MB + (96u << 10));
    int*      maxidx  = (int*)(ws + 24 * MB + (96u << 10) + 16);

    float* qout    = out;                 // [16384*512]
    float* lossp   = out + QELEMS;        // scalar
    float* perpp   = out + QELEMS + 1;    // scalar
    float* idxout  = out + QELEMS + 2;    // [16384]

    vq_conv_x  <<<QELEMS / 256, 256, 0, stream>>>(x, xh);
    vq_conv_emb<<<KCODES,       256, 0, stream>>>(emb, eh, cvec);
    vq_init    <<<1, 32, 0, stream>>>(sumsq, maxidx);
    vq_argmin  <<<NROWS / (16 * 8), 256, LDS_BYTES, stream>>>(xh, eh, cvec,
                                                              indices);
    vq_gather  <<<NROWS, 256, 0, stream>>>(x, emb, indices, qout, idxout,
                                           sumsq, maxidx);
    vq_finalize<<<1, 1, 0, stream>>>(sumsq, maxidx, lossp, perpp);
}
